// MultiHeadAttention_27358941675960
// MI455X (gfx1250) — matrix operand order repaired and checked by reading, not compiled, not run on hardware
//
#include <hip/hip_runtime.h>
#include <hip/hip_bf16.h>

typedef _Float16 f16;
typedef __attribute__((ext_vector_type(16))) _Float16 v16h;
typedef __attribute__((ext_vector_type(8)))  float    v8f;

#define D_MODEL 2048
#define N_HEADS 16
#define D_HEAD  128
#define B_SZ    64
#define T_LEN   128
#define M_TOK   (B_SZ * T_LEN)   // 8192

#define WMMA(a,b,c) __builtin_amdgcn_wmma_f32_16x16x32_f16(false,(a),false,(b),(short)0,(c),false,false)

// ---------------------------------------------------------------------------
// Fragment loaders for V_WMMA_F32_16X16X32_F16 per CDNA5 ISA §7.12.2.
// A (16x32, MxK): lanes 0-15 row M=lane, K={0..7,16..23}; lanes 16-31 K={8..15,24..31};
// B (32x16, KxN): lane lo = column N; lanes 0-15 K={0..7,16..23}; lanes 16-31 K={8..15,24..31}.
// Tiles are stored row-contiguous-in-K, so pairs of halves load as aligned dwords.
// ---------------------------------------------------------------------------
__device__ __forceinline__ v16h frag_ld_a(const f16* base, int stride, int row0) {
  int lane = threadIdx.x & 31;
  int lo = lane & 15, hi = (lane >> 4) & 1;
  union { v16h v; unsigned u[8]; } f;
  const f16* p = base + (size_t)(row0 + lo) * stride + hi * 8;
#pragma unroll
  for (int j = 0; j < 8; ++j) {
    int k = 2 * j + ((j >= 4) ? 8 : 0);
    f.u[j] = *(const unsigned*)(p + k);
  }
  return f.v;
}

__device__ __forceinline__ v16h frag_ld_b(const f16* base, int stride, int col0) {
  int lane = threadIdx.x & 31;
  int lo = lane & 15, hi = (lane >> 4) & 1;
  union { v16h v; unsigned u[8]; } f;
  const f16* p = base + (size_t)(col0 + lo) * stride + hi * 8;
#pragma unroll
  for (int j = 0; j < 8; ++j)
    f.u[j] = *(const unsigned*)(p + 2 * j + ((j >= 4) ? 8 : 0));
  return f.v;
}

// ---------------------------------------------------------------------------
__global__ void cvt_f32_f16(const float* __restrict__ s, f16* __restrict__ d, int n) {
  int i = blockIdx.x * 256 + threadIdx.x;
  if (i < n) d[i] = (f16)s[i];
}

// ---------------------------------------------------------------------------
// NT GEMM: Y[m,n] = sum_k X[m,k] * W[n,k];  M=8192, N=2048, K=2048.
// Block tile 128x64, 8 waves (4x2), each wave 32x32 (2x2 WMMA), K-step 32.
// Double-buffered LDS staging via GLOBAL_LOAD_ASYNC_TO_LDS_B128 (ASYNCcnt):
// per stage each thread issues 3 b128 async copies (2 for X tile, 1 for W tile).
// Row stride 40 halves (80B) keeps every 16B async write 16B-aligned.
// MODE 0: scatter f16 into [B, H, T, d_head];  MODE 1: f32 flat -> d_out.
// ---------------------------------------------------------------------------
template<int MODE>
__global__ __launch_bounds__(256) void gemm_nt(const f16* __restrict__ X,
                                               const f16* __restrict__ W,
                                               f16* __restrict__ Yh,
                                               float* __restrict__ Yf) {
  constexpr int STR = 40;
  __shared__ f16 Xs[2][128 * STR];
  __shared__ f16 Ws[2][64 * STR];
  const int tid = threadIdx.x;
  const int m0 = blockIdx.x << 7;
  const int n0 = blockIdx.y << 6;
  const int lane = tid & 31;
  const int lo = lane & 15, hi = (lane >> 4) & 1;
  const int wave = __builtin_amdgcn_readfirstlane(tid >> 5);  // scalar, uniform
  const int wm = wave >> 1, wn = wave & 1;

  // per-thread async-copy slots: X rows (2 ops) + W row (1 op), 16B chunks
  const int xr0 = tid >> 2,        xc = (tid & 3);   // X rows 0..63
  const int xr1 = 64 + (tid >> 2);                   // X rows 64..127
  const int wr  = tid >> 2;                          // W rows 0..63

  auto stage = [&](int k0, int buf) {
    unsigned gx0 = (unsigned)(((m0 + xr0) * D_MODEL + k0) * 2 + xc * 16);
    unsigned gx1 = (unsigned)(((m0 + xr1) * D_MODEL + k0) * 2 + xc * 16);
    unsigned gw  = (unsigned)(((n0 + wr ) * D_MODEL + k0) * 2 + xc * 16);
    unsigned lx0 = (unsigned)(uintptr_t)&Xs[buf][xr0 * STR + xc * 8];
    unsigned lx1 = (unsigned)(uintptr_t)&Xs[buf][xr1 * STR + xc * 8];
    unsigned lw  = (unsigned)(uintptr_t)&Ws[buf][wr  * STR + xc * 8];
    asm volatile("global_load_async_to_lds_b128 %0, %1, %2\n\t"
                 "global_load_async_to_lds_b128 %3, %4, %2\n\t"
                 "global_load_async_to_lds_b128 %5, %6, %7"
                 :
                 : "v"(lx0), "v"(gx0), "s"(X),
                   "v"(lx1), "v"(gx1),
                   "v"(lw),  "v"(gw),  "s"(W)
                 : "memory");
  };

  v8f z = {};
  v8f acc[2][2];
  acc[0][0] = z; acc[0][1] = z; acc[1][0] = z; acc[1][1] = z;

  stage(0, 0);
  const int NK = D_MODEL / 32;  // 64
  for (int kc = 0; kc < NK; ++kc) {
    const int cur = kc & 1;
    // own-wave async writes for buf[cur] have landed ...
    asm volatile("s_wait_asynccnt 0x0" ::: "memory");
    // ... and after the barrier, everyone's have; also nobody still reads buf[cur^1]
    __syncthreads();
    if (kc + 1 < NK) stage((kc + 1) * 32, cur ^ 1);   // DMA overlaps the WMMAs below

    const f16* xb = &Xs[cur][0];
    const f16* wb = &Ws[cur][0];
    v16h a0 = frag_ld_a(xb, STR, wm * 32);
    v16h a1 = frag_ld_a(xb, STR, wm * 32 + 16);
    v16h b0 = frag_ld_b(wb, STR, wn * 32);
    v16h b1 = frag_ld_b(wb, STR, wn * 32 + 16);
    acc[0][0] = WMMA(a0, b0, acc[0][0]);
    acc[0][1] = WMMA(a0, b1, acc[0][1]);
    acc[1][0] = WMMA(a1, b0, acc[1][0]);
    acc[1][1] = WMMA(a1, b1, acc[1][1]);
  }

#pragma unroll
  for (int ti = 0; ti < 2; ++ti)
#pragma unroll
    for (int tj = 0; tj < 2; ++tj)
#pragma unroll
      for (int r = 0; r < 8; ++r) {
        int m = m0 + wm * 32 + ti * 16 + r + 8 * hi;
        int d = n0 + wn * 32 + tj * 16 + lo;
        float v = acc[ti][tj][r];
        if (MODE == 0) {
          int b = m >> 7, t = m & 127, h = d >> 7, dh = d & 127;
          Yh[(((size_t)(b * N_HEADS + h) * T_LEN + t) << 7) + dh] = (f16)v;
        } else {
          Yf[(size_t)m * D_MODEL + d] = v;
        }
      }
}

// ---------------------------------------------------------------------------
// RoPE in place on Q and K, layout [B, H, T, d_head] f16; one thread per pair.
// ---------------------------------------------------------------------------
__global__ void rope_kernel(f16* __restrict__ Q, f16* __restrict__ K) {
  int i = blockIdx.x * 256 + threadIdx.x;
  const int total = B_SZ * N_HEADS * T_LEN * (D_HEAD / 2);  // 8,388,608
  if (i >= total) return;
  int ip = i & 63;
  int t  = (i >> 6) & 127;
  float theta = __expf(-0.143911568f * (float)ip);  // ln(10000)/64
  float ang = (float)t * theta;
  float sn, cs;
  __sincosf(ang, &sn, &cs);
  size_t base = (size_t)i * 2;
  float qe = (float)Q[base], qo = (float)Q[base + 1];
  Q[base]     = (f16)(qe * cs - qo * sn);
  Q[base + 1] = (f16)(qe * sn + qo * cs);
  float ke = (float)K[base], ko = (float)K[base + 1];
  K[base]     = (f16)(ke * cs - ko * sn);
  K[base + 1] = (f16)(ke * sn + ko * cs);
}

// ---------------------------------------------------------------------------
// Attention: one workgroup per (b,h). K and V^T staged in dynamic LDS.
// Wave-uniform causal-tile skipping is forced scalar via readfirstlane so
// WMMA never sits under EXEC-masked control flow.
// ---------------------------------------------------------------------------
__global__ __launch_bounds__(256) void attn_kernel(const f16* __restrict__ Q,
                                                   const f16* __restrict__ K,
                                                   const f16* __restrict__ V,
                                                   f16* __restrict__ A) {
  extern __shared__ f16 lds[];
  const int STR = 132;
  f16* Ks = lds;                 // [key][d], later reused as P [q][key]
  f16* Vt = lds + 128 * STR;     // [d][key]
  const int bh  = blockIdx.x;
  const int tid = threadIdx.x;
  const size_t off = (size_t)bh * (T_LEN * D_HEAD);
  const f16* Qp = Q + off;
  const f16* Kp = K + off;
  const f16* Vp = V + off;

  for (int i = tid; i < 128 * 64; i += 256) {        // K: dword copies
    int r = i >> 6, c = (i & 63) << 1;
    *(unsigned*)&Ks[r * STR + c] = *(const unsigned*)(Kp + r * 128 + c);
  }
  for (int i = tid; i < 128 * 128; i += 256) {       // V: transpose on store
    int r = i >> 7, d = i & 127;
    Vt[d * STR + r] = Vp[r * 128 + d];
  }
  __syncthreads();

  const int wave = __builtin_amdgcn_readfirstlane(tid >> 5);  // scalar, uniform
  const int lane = tid & 31, lo = lane & 15, hi = (lane >> 4) & 1;
  const int q0 = wave * 16;

  v16h aq[4];
#pragma unroll
  for (int s = 0; s < 4; ++s) aq[s] = frag_ld_a(Qp + s * 32, 128, q0);

  v8f sacc[8];
#pragma unroll
  for (int jt = 0; jt < 8; ++jt) {
    v8f c = {};
    if (jt <= wave) {          // scalar branch: EXEC stays all-1s for WMMA
#pragma unroll
      for (int s = 0; s < 4; ++s)
        c = WMMA(aq[s], frag_ld_b(Ks + s * 32, STR, jt * 16), c);
    }
    sacc[jt] = c;
  }

  const float scale = 0.08838834764831845f;  // 1/sqrt(128)
  float rmax[8], rsum[8], rinv[8];
#pragma unroll
  for (int r = 0; r < 8; ++r) rmax[r] = -3.0e38f;
#pragma unroll
  for (int jt = 0; jt < 8; ++jt) {
    if (jt > wave) continue;
#pragma unroll
    for (int r = 0; r < 8; ++r) {
      int row = q0 + r + 8 * hi;
      int col = jt * 16 + lo;
      float v = sacc[jt][r] * scale;
      if (col > row) v = -3.0e38f;               // causal mask
      sacc[jt][r] = v;
      rmax[r] = fmaxf(rmax[r], v);
    }
  }
#pragma unroll
  for (int r = 0; r < 8; ++r)
#pragma unroll
    for (int m = 1; m < 16; m <<= 1)
      rmax[r] = fmaxf(rmax[r], __shfl_xor(rmax[r], m, 32));
#pragma unroll
  for (int r = 0; r < 8; ++r) rsum[r] = 0.0f;
#pragma unroll
  for (int jt = 0; jt < 8; ++jt) {
    if (jt > wave) continue;
#pragma unroll
    for (int r = 0; r < 8; ++r) {
      float e = __expf(sacc[jt][r] - rmax[r]);
      sacc[jt][r] = e;
      rsum[r] += e;
    }
  }
#pragma unroll
  for (int r = 0; r < 8; ++r)
#pragma unroll
    for (int m = 1; m < 16; m <<= 1)
      rsum[r] += __shfl_xor(rsum[r], m, 32);
#pragma unroll
  for (int r = 0; r < 8; ++r) rinv[r] = 1.0f / rsum[r];

  __syncthreads();             // everyone finished reading Ks -> reuse for P
  f16* Ps = Ks;
#pragma unroll
  for (int jt = 0; jt < 8; ++jt)
#pragma unroll
    for (int r = 0; r < 8; ++r) {
      float pv = (jt <= wave) ? sacc[jt][r] * rinv[r] : 0.0f;
      Ps[(q0 + r + 8 * hi) * STR + jt * 16 + lo] = (f16)pv;
    }

  v8f oacc[8];
  v8f z = {};
#pragma unroll
  for (int dt = 0; dt < 8; ++dt) oacc[dt] = z;
#pragma unroll
  for (int s = 0; s < 4; ++s) {
    if (s > (wave >> 1)) continue;   // tail K-chunks are all-zero P (scalar)
    v16h ap = frag_ld_a(Ps + s * 32, STR, q0);
#pragma unroll
    for (int dt = 0; dt < 8; ++dt)
      oacc[dt] = WMMA(ap, frag_ld_b(Vt + s * 32, STR, dt * 16), oacc[dt]);
  }

  const int b = bh >> 4, h = bh & 15;
#pragma unroll
  for (int dt = 0; dt < 8; ++dt)
#pragma unroll
    for (int r = 0; r < 8; ++r) {
      int t = q0 + r + 8 * hi;
      int d = dt * 16 + lo;
      A[((size_t)(b * T_LEN + t)) * D_MODEL + h * D_HEAD + d] = (f16)oacc[dt][r];
    }
}

// ---------------------------------------------------------------------------
extern "C" void kernel_launch(void* const* d_in, const int* in_sizes, int n_in,
                              void* d_out, int out_size, void* d_ws, size_t ws_size,
                              hipStream_t stream) {
  const float* x  = (const float*)d_in[0];
  const float* Wq = (const float*)d_in[1];
  const float* Wk = (const float*)d_in[2];
  const float* Wv = (const float*)d_in[3];
  const float* Wo = (const float*)d_in[4];
  // d_in[5] = mask (causal, applied analytically)

  f16* ws16 = (f16*)d_ws;
  const size_t WSZ = (size_t)D_MODEL * D_MODEL;   // 4,194,304
  const size_t TSZ = (size_t)M_TOK * D_MODEL;     // 16,777,216
  f16* Wq16 = ws16;
  f16* Wk16 = ws16 + WSZ;
  f16* Wv16 = ws16 + 2 * WSZ;
  f16* Wo16 = ws16 + 3 * WSZ;
  f16* X16  = ws16 + 4 * WSZ;
  f16* Q16  = X16 + TSZ;
  f16* K16  = Q16 + TSZ;
  f16* V16  = K16 + TSZ;
  f16* A16  = V16 + TSZ;

  // 1) cast inputs to f16
  cvt_f32_f16<<<(int)((TSZ + 255) / 256), 256, 0, stream>>>(x,  X16, (int)TSZ);
  cvt_f32_f16<<<(int)((WSZ + 255) / 256), 256, 0, stream>>>(Wq, Wq16, (int)WSZ);
  cvt_f32_f16<<<(int)((WSZ + 255) / 256), 256, 0, stream>>>(Wk, Wk16, (int)WSZ);
  cvt_f32_f16<<<(int)((WSZ + 255) / 256), 256, 0, stream>>>(Wv, Wv16, (int)WSZ);
  cvt_f32_f16<<<(int)((WSZ + 255) / 256), 256, 0, stream>>>(Wo, Wo16, (int)WSZ);

  // 2) Q/K/V projections (scattered into [B,H,T,d_head])
  dim3 gg(M_TOK / 128, D_MODEL / 64);
  gemm_nt<0><<<gg, 256, 0, stream>>>(X16, Wq16, Q16, nullptr);
  gemm_nt<0><<<gg, 256, 0, stream>>>(X16, Wk16, K16, nullptr);
  gemm_nt<0><<<gg, 256, 0, stream>>>(X16, Wv16, V16, nullptr);

  // 3) RoPE on Q and K
  rope_kernel<<<32768, 256, 0, stream>>>(Q16, K16);

  // 4) causal attention, one workgroup per (b, h); 66KB dynamic LDS
  attn_kernel<<<B_SZ * N_HEADS, 256, 2 * 128 * 132 * (int)sizeof(f16), stream>>>(
      Q16, K16, V16, A16);

  // 5) output projection -> f32 d_out
  gemm_nt<1><<<gg, 256, 0, stream>>>(A16, Wo16, nullptr, (float*)d_out);
}
